// WorldModelBackbone_13125420057271
// MI455X (gfx1250) — matrix operand order repaired and checked by reading, not compiled, not run on hardware
//
#include <hip/hip_runtime.h>

// ---------------------------------------------------------------------------
// MI455X (gfx1250, wave32) world-model backbone.
// GEMMs: v_wmma_f32_16x16x32_bf16, 2x2 tiles per wave, A tile staged in LDS
// via global_load_async_to_lds_b128, in-register f32->bf16 conversion
// (weights are single-use per forward, so converting in registers is
// bandwidth-optimal). Attention: VALU flash kernel with LDS-staged K/V.
// ---------------------------------------------------------------------------

typedef __attribute__((ext_vector_type(16))) __bf16 bf16x16;
typedef __attribute__((ext_vector_type(8)))  float  floatx8;
typedef __attribute__((ext_vector_type(4)))  int    v4i;

#define DM      1024
#define HEADS   16
#define HDIM    64
#define TT      16
#define LTOKN   256
#define SS      262        // LTOK + 2 + NUM_REG
#define MTOT    (TT*SS)    // 4192 = 32*131
#define MLAT    (TT*LTOKN) // 4096
#define DHID    4096
#define INDIM   768
#define CTXW    9
#define APAD    68         // 64 K-floats + 4 pad -> conflict-free LDS frag reads

#if __has_builtin(__builtin_amdgcn_global_load_async_to_lds_b128)
#define USE_ASYNC_LDS 1
#else
#define USE_ASYNC_LDS 0
#endif

typedef __attribute__((address_space(1))) v4i* as1_v4i_ptr;
typedef __attribute__((address_space(3))) v4i* as3_v4i_ptr;

__device__ __forceinline__ void cvt4(bf16x16& d, int base, float4 f) {
    d[base + 0] = (__bf16)f.x;
    d[base + 1] = (__bf16)f.y;
    d[base + 2] = (__bf16)f.z;
    d[base + 3] = (__bf16)f.w;
}

// B-fragment: lane n = lane&15 reads W row (output col), K = grp*16 + 0..15.
__device__ __forceinline__ bf16x16 load_bfrag(const float* p) {
    const float4* q = (const float4*)p;
    float4 f0 = q[0], f1 = q[1], f2 = q[2], f3 = q[3];
    bf16x16 b;
    cvt4(b, 0, f0); cvt4(b, 4, f1); cvt4(b, 8, f2); cvt4(b, 12, f3);
    return b;
}

// A-fragment from the padded LDS tile: lane m = mbase + (lane&15),
// K = grp*16 + 0..15 (contiguous, matching the B fragment).
__device__ __forceinline__ bf16x16 lds_afrag(const float* As, int mbase,
                                             int kk, int n16, int grp) {
    const float* p = As + (mbase + n16) * APAD + kk + grp * 16;
    const float4* q = (const float4*)p;
    float4 f0 = q[0], f1 = q[1], f2 = q[2], f3 = q[3];
    bf16x16 a;
    cvt4(a, 0, f0); cvt4(a, 4, f1); cvt4(a, 8, f2); cvt4(a, 12, f3);
    return a;
}

// Stage 32 rows x 64 K of A into LDS (each thread: 2 x b128).
__device__ __forceinline__ void stage_a(const float* gsrc, float* ldst) {
#if USE_ASYNC_LDS
    __builtin_amdgcn_global_load_async_to_lds_b128(
        (as1_v4i_ptr)gsrc, (as3_v4i_ptr)ldst, 0, 0);
    __builtin_amdgcn_global_load_async_to_lds_b128(
        (as1_v4i_ptr)(gsrc + 4), (as3_v4i_ptr)(ldst + 4), 0, 0);
#if __has_builtin(__builtin_amdgcn_s_wait_asynccnt)
    __builtin_amdgcn_s_wait_asynccnt(0);
#else
    asm volatile("s_wait_asynccnt 0x0" ::: "memory");
#endif
#else
    *(float4*)ldst       = *(const float4*)gsrc;
    *(float4*)(ldst + 4) = *(const float4*)(gsrc + 4);
#endif
}

// ---------------------------------------------------------------------------
// WMMA GEMM: C[M,N] = A[M,K] * W[N,K]^T (+resid)(+bias), row remap optional.
// Block: 32 rows x 256 cols; wave w owns a 32x32 tile quad (2 m x 2 n).
// ---------------------------------------------------------------------------
template<int BIAS, int RESID, int REMAP>
__global__ __launch_bounds__(256) void k_gemm(
    const float* __restrict__ A, const float* __restrict__ W,
    float* __restrict__ C, const float* __restrict__ resid,
    const float* __restrict__ bias, int M, int N, int K)
{
    __shared__ float As[32 * APAD];
    const int tid  = threadIdx.x;
    const int wave = tid >> 5;
    const int lane = tid & 31;
    const int tmb  = blockIdx.y * 32;
    const int tn0  = blockIdx.x * 256 + wave * 32;
    if (tn0 >= N) return;                       // wave-uniform
    const int n16 = lane & 15, grp = lane >> 4;

    const int crow = tid >> 3;                  // 0..31
    const int ccol = (tid & 7) * 8;             // 0..56
    const float* gsrc = A + (size_t)(tmb + crow) * K + ccol;
    float* ldst = As + crow * APAD + ccol;

    const float* pb0 = W + (size_t)(tn0 + n16) * K + grp * 16;
    const float* pb1 = W + (size_t)(tn0 + 16 + n16) * K + grp * 16;

    floatx8 acc00 = {0.f,0.f,0.f,0.f,0.f,0.f,0.f,0.f};
    floatx8 acc01 = acc00, acc10 = acc00, acc11 = acc00;

    for (int k0 = 0; k0 < K; k0 += 64) {
        stage_a(gsrc + k0, ldst);
        __builtin_prefetch(pb0 + k0 + 128, 0, 0);
        __builtin_prefetch(pb1 + k0 + 128, 0, 0);
        __syncthreads();
        #pragma unroll
        for (int kk = 0; kk < 64; kk += 32) {
            bf16x16 a0 = lds_afrag(As, 0,  kk, n16, grp);
            bf16x16 a1 = lds_afrag(As, 16, kk, n16, grp);
            bf16x16 b0 = load_bfrag(pb0 + k0 + kk);
            bf16x16 b1 = load_bfrag(pb1 + k0 + kk);
            acc00 = __builtin_amdgcn_wmma_f32_16x16x32_bf16(
                        false, a0, false, b0, (short)0, acc00, false, false);
            acc01 = __builtin_amdgcn_wmma_f32_16x16x32_bf16(
                        false, a0, false, b1, (short)0, acc01, false, false);
            acc10 = __builtin_amdgcn_wmma_f32_16x16x32_bf16(
                        false, a1, false, b0, (short)0, acc10, false, false);
            acc11 = __builtin_amdgcn_wmma_f32_16x16x32_bf16(
                        false, a1, false, b1, (short)0, acc11, false, false);
        }
        __syncthreads();
    }

    floatx8* accs[4] = { &acc00, &acc01, &acc10, &acc11 };
    #pragma unroll
    for (int t = 0; t < 4; ++t) {
        const int tm = tmb + (t >> 1) * 16;
        const int col = tn0 + (t & 1) * 16 + n16;
        floatx8& acc = *accs[t];
        #pragma unroll
        for (int i = 0; i < 8; ++i) {
            int row = tm + i + 8 * grp;          // C layout: VGPR i, half grp
            size_t orow = REMAP ? ((size_t)(row >> 8) * SS + 6 + (row & 255))
                                : (size_t)row;
            float v = acc[i];
            if (BIAS)  v += bias[col];
            if (RESID) v += resid[orow * (size_t)N + col];
            C[orow * (size_t)N + col] = v;
        }
    }
}

// ---------------------------------------------------------------------------
// Gated WMMA GEMM (SwiGLU): out = silu(A@W1^T) * (A@W2^T), W2 = W rows +Nh.
// Block: 32 rows x 128 cols; wave owns 2 m-tiles x 1 n-tile x 2 gate halves.
// ---------------------------------------------------------------------------
__global__ __launch_bounds__(256) void k_gemm_gated(
    const float* __restrict__ A, const float* __restrict__ W,
    float* __restrict__ Out, int M, int Nh, int K)
{
    __shared__ float As[32 * APAD];
    const int tid  = threadIdx.x;
    const int wave = tid >> 5;
    const int lane = tid & 31;
    const int tmb  = blockIdx.y * 32;
    const int tn0  = blockIdx.x * 128 + wave * 16;
    if (tn0 >= Nh) return;
    const int n16 = lane & 15, grp = lane >> 4;

    const int crow = tid >> 3;
    const int ccol = (tid & 7) * 8;
    const float* gsrc = A + (size_t)(tmb + crow) * K + ccol;
    float* ldst = As + crow * APAD + ccol;

    const float* pb1 = W + (size_t)(tn0 + n16) * K + grp * 16;        // g1
    const float* pb2 = W + (size_t)(tn0 + Nh + n16) * K + grp * 16;   // g2

    floatx8 acc00 = {0.f,0.f,0.f,0.f,0.f,0.f,0.f,0.f};
    floatx8 acc01 = acc00, acc10 = acc00, acc11 = acc00;

    for (int k0 = 0; k0 < K; k0 += 64) {
        stage_a(gsrc + k0, ldst);
        __builtin_prefetch(pb1 + k0 + 128, 0, 0);
        __builtin_prefetch(pb2 + k0 + 128, 0, 0);
        __syncthreads();
        #pragma unroll
        for (int kk = 0; kk < 64; kk += 32) {
            bf16x16 a0 = lds_afrag(As, 0,  kk, n16, grp);
            bf16x16 a1 = lds_afrag(As, 16, kk, n16, grp);
            bf16x16 b1 = load_bfrag(pb1 + k0 + kk);
            bf16x16 b2 = load_bfrag(pb2 + k0 + kk);
            acc00 = __builtin_amdgcn_wmma_f32_16x16x32_bf16(
                        false, a0, false, b1, (short)0, acc00, false, false);
            acc01 = __builtin_amdgcn_wmma_f32_16x16x32_bf16(
                        false, a0, false, b2, (short)0, acc01, false, false);
            acc10 = __builtin_amdgcn_wmma_f32_16x16x32_bf16(
                        false, a1, false, b1, (short)0, acc10, false, false);
            acc11 = __builtin_amdgcn_wmma_f32_16x16x32_bf16(
                        false, a1, false, b2, (short)0, acc11, false, false);
        }
        __syncthreads();
    }

    const int col = tn0 + n16;
    #pragma unroll
    for (int i = 0; i < 8; ++i) {
        int r0 = tmb + i + 8 * grp;
        int r1 = r0 + 16;
        float g1 = acc00[i], g2 = acc01[i];
        Out[(size_t)r0 * Nh + col] = (g1 / (1.0f + __expf(-g1))) * g2;
        float h1 = acc10[i], h2 = acc11[i];
        Out[(size_t)r1 * Nh + col] = (h1 / (1.0f + __expf(-h1))) * h2;
    }
}

// ---------------------------------------------------------------------------
// RMSNorm; gather=1 compacts latent tokens (x row t*262+6+lt -> y row t*256+lt)
// ---------------------------------------------------------------------------
__global__ __launch_bounds__(256) void k_rms(
    const float* __restrict__ x, const float* __restrict__ w,
    float* __restrict__ y, int gather)
{
    const int b = blockIdx.x;
    const size_t srow = gather ? ((size_t)(b >> 8) * SS + 6 + (b & 255))
                               : (size_t)b;
    const float* xr = x + srow * DM;
    const int tid = threadIdx.x;

    float v[4], ss = 0.f;
    #pragma unroll
    for (int j = 0; j < 4; ++j) { v[j] = xr[tid + j * 256]; ss += v[j] * v[j]; }
    #pragma unroll
    for (int m = 16; m >= 1; m >>= 1) ss += __shfl_xor(ss, m, 32);
    __shared__ float red[8];
    if ((tid & 31) == 0) red[tid >> 5] = ss;
    __syncthreads();
    float tot = 0.f;
    #pragma unroll
    for (int j = 0; j < 8; ++j) tot += red[j];
    const float sc = rsqrtf(tot * (1.0f / DM) + 1e-6f);

    float* yr = y + (size_t)b * DM;
    #pragma unroll
    for (int j = 0; j < 4; ++j)
        yr[tid + j * 256] = v[j] * sc * w[tid + j * 256];
}

// ---------------------------------------------------------------------------
// Attention: one block per (group g, head h); token i row = g*gstride+i*istride.
// Lane owns dims (lane, lane+32) so the RoPE pair is in-lane and the RMS
// reduce is a wave32 shfl_xor tree. Online softmax, 50*tanh(z/50) softcap.
// mode 0: spatial mask (q==0 attends only k==0); mode 1: causal window CTXW.
// ---------------------------------------------------------------------------
__global__ __launch_bounds__(256) void k_attn(
    const float* __restrict__ qkv, float* __restrict__ o,
    const float* __restrict__ qg, const float* __restrict__ kg,
    int L, int gstride, int istride, int mode)
{
    extern __shared__ float smem[];
    float* Ks = smem;
    float* Vs = smem + (size_t)L * HDIM;

    const int g = blockIdx.x, h = blockIdx.y;
    const int tid = threadIdx.x;
    const int wave = tid >> 5, lane = tid & 31;

    const float qg1 = qg[lane], qg2 = qg[lane + 32];
    const float kg1 = kg[lane], kg2 = kg[lane + 32];
    const float inv = __expf(-(float)lane * 0.28782313662425574f); // ln(1e4)/32

    for (int i = wave; i < L; i += 8) {
        size_t r = (size_t)g * gstride + (size_t)i * istride;
        const float* kp = qkv + r * (3 * DM) + DM + h * HDIM;
        const float* vp = qkv + r * (3 * DM) + 2 * DM + h * HDIM;
        float k1 = kp[lane], k2 = kp[lane + 32];
        float ss = k1 * k1 + k2 * k2;
        #pragma unroll
        for (int m = 16; m >= 1; m >>= 1) ss += __shfl_xor(ss, m, 32);
        float sc = rsqrtf(ss * (1.0f / HDIM) + 1e-6f);
        k1 *= sc * kg1; k2 *= sc * kg2;
        float ang = (float)i * inv;
        float c = __cosf(ang), s = __sinf(ang);
        Ks[i * HDIM + lane]      = k1 * c - k2 * s;
        Ks[i * HDIM + lane + 32] = k1 * s + k2 * c;
        Vs[i * HDIM + lane]      = vp[lane];
        Vs[i * HDIM + lane + 32] = vp[lane + 32];
    }
    __syncthreads();

    for (int q = wave; q < L; q += 8) {
        size_t r = (size_t)g * gstride + (size_t)q * istride;
        const float* qp = qkv + r * (3 * DM) + h * HDIM;
        float q1 = qp[lane], q2 = qp[lane + 32];
        float ss = q1 * q1 + q2 * q2;
        #pragma unroll
        for (int m = 16; m >= 1; m >>= 1) ss += __shfl_xor(ss, m, 32);
        float sc = rsqrtf(ss * (1.0f / HDIM) + 1e-6f);
        q1 *= sc * qg1; q2 *= sc * qg2;
        float ang = (float)q * inv;
        float c = __cosf(ang), s = __sinf(ang);
        float qa = q1 * c - q2 * s;
        float qb = q1 * s + q2 * c;

        float mx = -1e38f, sum = 0.f, o1 = 0.f, o2 = 0.f;
        for (int kk = 0; kk < L; ++kk) {
            float d = qa * Ks[kk * HDIM + lane] + qb * Ks[kk * HDIM + lane + 32];
            #pragma unroll
            for (int m = 16; m >= 1; m >>= 1) d += __shfl_xor(d, m, 32);
            float z = 50.0f * tanhf(d * 0.125f * 0.02f);
            bool ok = (mode == 0) ? ((q != 0) || (kk == 0))
                                  : ((kk <= q) && (kk >= q - CTXW));
            if (!ok) z = -1e30f;
            float nm  = fmaxf(mx, z);
            float scl = __expf(mx - nm);
            float p   = __expf(z - nm);
            sum = sum * scl + p;
            o1  = o1 * scl + p * Vs[kk * HDIM + lane];
            o2  = o2 * scl + p * Vs[kk * HDIM + lane + 32];
            mx = nm;
        }
        float is = 1.0f / sum;
        float* op = o + r * DM + h * HDIM;
        op[lane]      = o1 * is;
        op[lane + 32] = o2 * is;
    }
}

// ---------------------------------------------------------------------------
// Timestep-embedding MLP: sig[t,:] = silu(emb(t) @ W1^T + b1) @ W2^T + b2
// ---------------------------------------------------------------------------
__global__ __launch_bounds__(256) void k_sig_mlp(
    const float* __restrict__ sigl,
    const float* __restrict__ w1, const float* __restrict__ b1,
    const float* __restrict__ w2, const float* __restrict__ b2,
    float* __restrict__ sig)
{
    __shared__ float emb[256];
    __shared__ float mid[DM];
    const int t = blockIdx.x, tid = threadIdx.x;
    const float tv = sigl[t];

    float fj = __expf(-(float)(tid & 127) * 0.07195578415606393f); // ln(1e4)/128
    float arg = tv * fj;
    emb[tid] = (tid < 128) ? __cosf(arg) : __sinf(arg);
    __syncthreads();

    for (int o = tid; o < DM; o += 256) {
        float acc = b1[o];
        const float* wr = w1 + (size_t)o * 256;
        for (int k = 0; k < 256; ++k) acc += emb[k] * wr[k];
        mid[o] = acc / (1.0f + __expf(-acc));
    }
    __syncthreads();

    for (int o = tid; o < DM; o += 256) {
        float acc = b2[o];
        const float* wr = w2 + (size_t)o * DM;
        for (int k = 0; k < DM; ++k) acc += mid[k] * wr[k];
        sig[(size_t)t * DM + o] = acc;
    }
}

// ---------------------------------------------------------------------------
// Assemble x rows s=0..5 per frame: [sig | act | reg*4].
// ---------------------------------------------------------------------------
__global__ __launch_bounds__(256) void k_assemble(
    const float* __restrict__ sig, const float* __restrict__ actions,
    const float* __restrict__ base, const float* __restrict__ apw,
    const float* __restrict__ apb, const float* __restrict__ regs,
    float* __restrict__ x)
{
    const int t = blockIdx.x, tid = threadIdx.x;
    float* xr = x + (size_t)t * SS * DM;
    for (int c = tid; c < DM; c += 256) {
        xr[c] = sig[(size_t)t * DM + c];
        float a = base[c] + apb[c];
        const float* wr = apw + (size_t)c * 8;
        #pragma unroll
        for (int k = 0; k < 8; ++k) a += actions[t * 8 + k] * wr[k];
        xr[DM + c] = a;
        #pragma unroll
        for (int rg = 0; rg < 4; ++rg)
            xr[(2 + rg) * DM + c] = regs[(size_t)rg * DM + c];
    }
}

// ---------------------------------------------------------------------------
extern "C" void kernel_launch(void* const* d_in, const int* in_sizes, int n_in,
                              void* d_out, int out_size, void* d_ws, size_t ws_size,
                              hipStream_t stream)
{
    (void)in_sizes; (void)n_in; (void)out_size; (void)ws_size;
    const float* noisy = (const float*)d_in[0];
    const float* sigl  = (const float*)d_in[1];
    const float* acts  = (const float*)d_in[2];
    const float* ipw   = (const float*)d_in[3];
    const float* nw1   = (const float*)d_in[4];
    const float* nb1   = (const float*)d_in[5];
    const float* nw2   = (const float*)d_in[6];
    const float* nb2   = (const float*)d_in[7];
    const float* bact  = (const float*)d_in[8];
    const float* apw   = (const float*)d_in[9];
    const float* apb   = (const float*)d_in[10];
    const float* regs  = (const float*)d_in[11];
    const float* ln1   = (const float*)d_in[12];
    const float* swqkv = (const float*)d_in[13];
    const float* swo   = (const float*)d_in[14];
    const float* sqg   = (const float*)d_in[15];
    const float* skg   = (const float*)d_in[16];
    const float* lnt   = (const float*)d_in[17];
    const float* twqkv = (const float*)d_in[18];
    const float* two   = (const float*)d_in[19];
    const float* tqg   = (const float*)d_in[20];
    const float* tkg   = (const float*)d_in[21];
    const float* ln2   = (const float*)d_in[22];
    const float* w12   = (const float*)d_in[23];
    const float* w3    = (const float*)d_in[24];
    const float* fnw   = (const float*)d_in[25];
    const float* outw  = (const float*)d_in[26];
    const float* outb  = (const float*)d_in[27];
    float* out = (float*)d_out;

    // Workspace carve-out (f32), ~190 MB.
    float* ws = (float*)d_ws;
    size_t off = 0;
    float* x    = ws + off; off += (size_t)MTOT * DM;
    float* h    = ws + off; off += (size_t)MTOT * DM;
    float* qkv  = ws + off; off += (size_t)MTOT * 3 * DM;
    float* o    = ws + off; off += (size_t)MTOT * DM;
    float* aw   = ws + off; off += (size_t)MTOT * DHID;
    float* sig  = ws + off; off += (size_t)TT * DM;
    float* hfin = ws + off; off += (size_t)MLAT * DM;

    const size_t shm_sp = 2 * (size_t)SS * HDIM * sizeof(float); // 134 KB
    const size_t shm_tp = 2 * (size_t)TT * HDIM * sizeof(float);

    // --- Embedding ---
    k_sig_mlp<<<TT, 256, 0, stream>>>(sigl, nw1, nb1, nw2, nb2, sig);
    k_assemble<<<TT, 256, 0, stream>>>(sig, acts, bact, apw, apb, regs, x);
    k_gemm<0,0,1><<<dim3(DM / 256, MLAT / 32), 256, 0, stream>>>(
        noisy, ipw, x, nullptr, nullptr, MLAT, DM, INDIM);

    // --- Transformer layers ---
    for (int l = 0; l < 8; ++l) {
        k_rms<<<MTOT, 256, 0, stream>>>(x, ln1 + (size_t)l * DM, h, 0);
        k_gemm<0,0,0><<<dim3(3 * DM / 256, MTOT / 32), 256, 0, stream>>>(
            h, swqkv + (size_t)l * 3 * DM * DM, qkv, nullptr, nullptr,
            MTOT, 3 * DM, DM);
        k_attn<<<dim3(TT, HEADS), 256, shm_sp, stream>>>(
            qkv, o, sqg + (size_t)l * HDIM, skg + (size_t)l * HDIM,
            SS, SS, 1, 0);
        k_gemm<0,1,0><<<dim3(DM / 256, MTOT / 32), 256, 0, stream>>>(
            o, swo + (size_t)l * DM * DM, x, x, nullptr, MTOT, DM, DM);

        if ((l & 3) == 0) {   // temporal block every INTERVAL layers
            k_rms<<<MTOT, 256, 0, stream>>>(x, lnt + (size_t)l * DM, h, 0);
            k_gemm<0,0,0><<<dim3(3 * DM / 256, MTOT / 32), 256, 0, stream>>>(
                h, twqkv + (size_t)l * 3 * DM * DM, qkv, nullptr, nullptr,
                MTOT, 3 * DM, DM);
            k_attn<<<dim3(SS, HEADS), 256, shm_tp, stream>>>(
                qkv, o, tqg + (size_t)l * HDIM, tkg + (size_t)l * HDIM,
                TT, 1, SS, 1);
            k_gemm<0,1,0><<<dim3(DM / 256, MTOT / 32), 256, 0, stream>>>(
                o, two + (size_t)l * DM * DM, x, x, nullptr, MTOT, DM, DM);
        }

        k_rms<<<MTOT, 256, 0, stream>>>(x, ln2 + (size_t)l * DM, h, 0);
        k_gemm_gated<<<dim3(DHID / 128, MTOT / 32), 256, 0, stream>>>(
            h, w12 + (size_t)l * 2 * DHID * DM, aw, MTOT, DHID, DM);
        k_gemm<0,1,0><<<dim3(DM / 256, MTOT / 32), 256, 0, stream>>>(
            aw, w3 + (size_t)l * DM * DHID, x, x, nullptr, MTOT, DM, DHID);
    }

    // --- Output head ---
    k_rms<<<MLAT, 256, 0, stream>>>(x, fnw, hfin, 1);
    k_gemm<1,0,0><<<dim3(INDIM / 256, MLAT / 32), 256, 0, stream>>>(
        hfin, outw, out, nullptr, outb, MLAT, INDIM, DM);
}
